// NeuralPhysicsEngine_37117107372493
// MI455X (gfx1250) — matrix operand order repaired and checked by reading, not compiled, not run on hardware
//
#include <hip/hip_runtime.h>
#include <hip/hip_bf16.h>
#include <stdint.h>

// ---------------------------------------------------------------------------
// NeuralPhysicsEngine forward for MI455X (gfx1250, wave32, WMMA)
//
// Math restructuring (exact):
//   pair@W_pair and enc_W0 both decompose per-node (no relu before them):
//     gf[n] = ([prev_n,cur_n] @ Wp[:16]) @ enc_W0    (50)
//     gc[n] = ([prev_n,cur_n] @ Wp[16:]) @ enc_W0    (50)
//   Per pair:  h1 = relu(gf[f] + gc[c]);  4x (50x50 relu GEMM);  mask-sum.
// The 4-layer 50x50 chain runs on v_wmma_f32_16x16x32_f16:
//   M = 16 focus rows, K = 64 (50 zero-padded), N = 4 tiles of 16.
// enc_W1..4 B-fragments (256 VGPRs) stay register-resident across the loop.
// ---------------------------------------------------------------------------

typedef __attribute__((ext_vector_type(16))) _Float16 v16h;
typedef __attribute__((ext_vector_type(8)))  _Float16 v8h;
typedef __attribute__((ext_vector_type(8)))  float    v8f;

#define B_      4
#define N_      512
#define D_      8
#define H_      50      // hidden width
#define HP      64      // padded hidden width (K and N pad)
#define NT      4       // N-tiles of 16 covering HP
#define NL      4       // enc layers done via WMMA (enc_W1..enc_W4)
#define WAVES   8
#define CPW     (N_/WAVES) // contexts per wave

__device__ __forceinline__ float relu_f(float x) { return x > 0.f ? x : 0.f; }

// Async global->LDS staging (gfx1250, tracked by ASYNCcnt).
// VDST VGPR = LDS byte offset, VADDR = 64-bit global address.
__device__ __forceinline__ void async_ld_b32(void* lds_generic, const void* gptr)
{
    uint32_t loff = (uint32_t)(uintptr_t)(__attribute__((address_space(3))) char*)lds_generic;
    asm volatile("global_load_async_to_lds_b32 %0, %1, off"
                 :: "v"(loff), "v"(gptr) : "memory");
}
__device__ __forceinline__ void wait_asynccnt0()
{
#if __has_builtin(__builtin_amdgcn_s_wait_asynccnt)
    __builtin_amdgcn_s_wait_asynccnt(0);
#else
    asm volatile("s_wait_asynccnt 0" ::: "memory");
#endif
}

// ---------------------------------------------------------------------------
// Prep 1: pack enc_W1..4 (50x50, row-major) into per-lane WMMA B fragments.
// B layout used here for 16-bit 32x16 (contiguous K, not ISA table order): lane L holds column N = L&15,
// K = 16*(L>>4) + e for fragment element e (kstep adds 32).  Pads -> 0.
// ---------------------------------------------------------------------------
__global__ void pack_weights_kernel(const float* __restrict__ w1,
                                    const float* __restrict__ w2,
                                    const float* __restrict__ w3,
                                    const float* __restrict__ w4,
                                    _Float16* __restrict__ wfrag)
{
    int tid = blockIdx.x * blockDim.x + threadIdx.x;     // 0..1023
    if (tid >= NL * NT * 2 * 32) return;
    int lane  = tid & 31;
    int kstep = (tid >> 5) & 1;
    int nt    = (tid >> 6) & 3;
    int l     = (tid >> 8) & 3;
    const float* W = (l == 0) ? w1 : (l == 1) ? w2 : (l == 2) ? w3 : w4;
    int base  = ((((l * NT + nt) * 2 + kstep) * 32) + lane) * 16;
    int kbase = 32 * kstep + 16 * (lane >> 4);
    int n     = 16 * nt + (lane & 15);
#pragma unroll
    for (int e = 0; e < 16; ++e) {
        int k   = kbase + e;
        float v = (k < H_ && n < H_) ? W[k * H_ + n] : 0.f;
        wfrag[base + e] = (_Float16)v;
    }
}

// ---------------------------------------------------------------------------
// Prep 2: per-node projections gf/gc (fold W_pair halves and enc_W0).
// ---------------------------------------------------------------------------
__global__ void node_proj_kernel(const float* __restrict__ states,
                                 const float* __restrict__ Wp,   // (32,25)
                                 const float* __restrict__ W0,   // (25,50)
                                 float* __restrict__ gf,
                                 float* __restrict__ gc)
{
    int node = blockIdx.x * blockDim.x + threadIdx.x;
    if (node >= B_ * N_) return;
    int b = node >> 9, n = node & (N_ - 1);
    const float* prev = states + ((size_t)(b * 3 + 0) * N_ + n) * D_;
    const float* cur  = states + ((size_t)(b * 3 + 1) * N_ + n) * D_;
    float feat[16];
#pragma unroll
    for (int k = 0; k < 8; ++k) { feat[k] = prev[k]; feat[8 + k] = cur[k]; }
    float af[25], ac[25];
#pragma unroll
    for (int o = 0; o < 25; ++o) {
        float sf = 0.f, sc = 0.f;
#pragma unroll
        for (int k = 0; k < 16; ++k) {
            sf += feat[k] * Wp[k * 25 + o];
            sc += feat[k] * Wp[(16 + k) * 25 + o];
        }
        af[o] = sf; ac[o] = sc;
    }
    for (int j = 0; j < H_; ++j) {
        float sf = 0.f, sc = 0.f;
#pragma unroll
        for (int o = 0; o < 25; ++o) {
            sf += af[o] * W0[o * H_ + j];
            sc += ac[o] * W0[o * H_ + j];
        }
        gf[node * H_ + j] = sf;
        gc[node * H_ + j] = sc;
    }
}

// ---------------------------------------------------------------------------
// Main: pairwise encoder on WMMA.  Block = (b, 16-focus tile); 8 waves split
// the 512 contexts.  enc weights live in 256 VGPRs for the whole loop.
// Each wave writes its own 16x50 partial sum (no atomics).
// ---------------------------------------------------------------------------
__global__ void __launch_bounds__(256, 1)
pair_enc_kernel(const float* __restrict__ states,
                const float* __restrict__ gf_g,
                const float* __restrict__ gc_g,
                const _Float16* __restrict__ wfrag_g,
                float* __restrict__ partial)
{
    __shared__ float   gf_lds[16][H_];                              // 3.2 KB
    __shared__ float   cpos[N_ * 2];                                // 4 KB
    __shared__ __align__(16) _Float16 hb_all[WAVES][16 * HP];       // 16 KB
    __shared__ float   gcb_all[WAVES][H_];                          // 1.6 KB
    __shared__ float   maskb_all[WAVES][16];                        // 0.5 KB

    const int tid   = threadIdx.x;
    const int blk   = blockIdx.x;          // 0..127
    const int b     = blk >> 5;
    const int fbase = (blk & 31) * 16;

    // async staging of block-shared data (gfx1250 ASYNC path)
    {
        const float* curbase = states + (size_t)(b * 3 + 1) * N_ * D_;
        for (int i = tid; i < N_ * 2; i += 256) {
            int n = i >> 1, d = i & 1;
            async_ld_b32(&cpos[i], curbase + n * D_ + d);
        }
        for (int i = tid; i < 16 * H_; i += 256) {
            int r = i / H_, k = i % H_;
            async_ld_b32(&gf_lds[r][k], gf_g + ((size_t)(b * N_) + fbase + r) * H_ + k);
        }
        wait_asynccnt0();
    }
    __syncthreads();

    const int wv   = tid >> 5;
    const int lane = tid & 31;
    _Float16* hb    = hb_all[wv];
    float*    gcb   = gcb_all[wv];
    float*    maskb = maskb_all[wv];

    const int m  = lane & 15;         // A row / D column within tile
    const int h8 = (lane >> 4) * 8;   // lane-group K/row offset

    // Register-resident B fragments: 4 layers x 4 n-tiles x 2 k-steps.
    v16h Wb[NL][NT][2];
#pragma unroll
    for (int l = 0; l < NL; ++l)
#pragma unroll
        for (int nt = 0; nt < NT; ++nt)
#pragma unroll
            for (int ks = 0; ks < 2; ++ks)
                Wb[l][nt][ks] =
                    *(const v16h*)&wfrag_g[((((l * NT + nt) * 2 + ks) * 32) + lane) * 16];

    float fpx = 0.f, fpy = 0.f;
    if (lane < 16) { fpx = cpos[2 * (fbase + lane)]; fpy = cpos[2 * (fbase + lane) + 1]; }

    v8f acc[NT] = {};                 // masked enc_sum accumulators (f32)

    const int c0 = wv * CPW;
    for (int c = c0; c < c0 + CPW; ++c) {
        // mask row for each of the 16 focus rows (squared-distance test)
        bool p = false;
        if (lane < 16) {
            float dx = fpx - cpos[2 * c];
            float dy = fpy - cpos[2 * c + 1];
            float d2 = dx * dx + dy * dy;
            float mv = (d2 > 0.f && d2 < 0.25f) ? 1.f : 0.f;
            maskb[lane] = mv;
            p = (mv > 0.f);
        }
#if __has_builtin(__builtin_amdgcn_ballot_w32)
        if (__builtin_amdgcn_ballot_w32(p) == 0u) continue;   // uniform skip
#endif
        // stage gc[c] (50 floats)
        if (lane < 25) {
            gcb[lane]      = gc_g[((size_t)(b * N_) + c) * H_ + lane];
            gcb[lane + 25] = gc_g[((size_t)(b * N_) + c) * H_ + lane + 25];
        }
        // h1 = relu(gf[f] + gc[c]) into the 16x64 f16 activation tile
        {
            int row = lane >> 1;
            int kb  = (lane & 1) * 32;
#pragma unroll
            for (int e = 0; e < 32; ++e) {
                int k = kb + e;
                float v = 0.f;
                if (k < H_) v = relu_f(gf_lds[row][k] + gcb[k]);
                hb[row * HP + k] = (_Float16)v;
            }
        }
        float mrow[8];
#pragma unroll
        for (int r = 0; r < 8; ++r) mrow[r] = maskb[h8 + r];

#pragma unroll
        for (int l = 0; l < NL; ++l) {
            // A fragments: 16-bit A 16x32, contiguous K order (matches B packing):
            //   lane group g: element e -> K=16g+e
            v8h lo0 = *(const v8h*)&hb[m * HP +  0 + 2 * h8];
            v8h hi0 = *(const v8h*)&hb[m * HP +  8 + 2 * h8];
            v8h lo1 = *(const v8h*)&hb[m * HP + 32 + 2 * h8];
            v8h hi1 = *(const v8h*)&hb[m * HP + 40 + 2 * h8];
            v16h a0 = __builtin_shufflevector(lo0, hi0, 0,1,2,3,4,5,6,7,8,9,10,11,12,13,14,15);
            v16h a1 = __builtin_shufflevector(lo1, hi1, 0,1,2,3,4,5,6,7,8,9,10,11,12,13,14,15);
#pragma unroll
            for (int nt = 0; nt < NT; ++nt) {
                v8f cz = {};
                v8f d0 = __builtin_amdgcn_wmma_f32_16x16x32_f16(
                    false, a0, false, Wb[l][nt][0], (short)0, cz, false, false);
                d0 = __builtin_amdgcn_wmma_f32_16x16x32_f16(
                    false, a1, false, Wb[l][nt][1], (short)0, d0, false, false);
                if (l < NL - 1) {
                    // D layout: row = h8+r, col = (lane&15)+16*nt
#pragma unroll
                    for (int r = 0; r < 8; ++r) {
                        float v = relu_f(d0[r]);
                        hb[(h8 + r) * HP + (lane & 15) + 16 * nt] = (_Float16)v;
                    }
                } else {
#pragma unroll
                    for (int r = 0; r < 8; ++r)
                        acc[nt][r] += mrow[r] * relu_f(d0[r]);
                }
            }
        }
    }

    // per-wave partial sums: partial[(blk*8+wv)][row 0..15][col 0..49]
    float* pw = partial + (size_t)((blk * WAVES + wv) * 16) * H_;
#pragma unroll
    for (int nt = 0; nt < NT; ++nt) {
        int col = (lane & 15) + 16 * nt;
        if (col < H_) {
#pragma unroll
            for (int r = 0; r < 8; ++r)
                pw[(h8 + r) * H_ + col] = acc[nt][r];
        }
    }
}

// ---------------------------------------------------------------------------
// Decoder: one thread per node; sums the 8 wave-partials, runs the small MLP.
// ---------------------------------------------------------------------------
__global__ void decoder_kernel(const float* __restrict__ states,
                               const float* __restrict__ partial,
                               const float* __restrict__ W0, const float* __restrict__ b0,
                               const float* __restrict__ W1, const float* __restrict__ b1,
                               const float* __restrict__ W2, const float* __restrict__ b2,
                               const float* __restrict__ W3, const float* __restrict__ b3,
                               const float* __restrict__ W4, const float* __restrict__ b4,
                               float* __restrict__ out)
{
    int node = blockIdx.x * blockDim.x + threadIdx.x;
    if (node >= B_ * N_) return;
    int b = node >> 9, n = node & (N_ - 1);
    int ftile = n >> 4, row = n & 15;
    int blk = b * 32 + ftile;

    float y[66];
    for (int j = 0; j < H_; ++j) {
        float s = 0.f;
#pragma unroll
        for (int w = 0; w < WAVES; ++w)
            s += partial[(size_t)((blk * WAVES + w) * 16 + row) * H_ + j];
        y[j] = s;
    }
    const float* prev = states + ((size_t)(b * 3 + 0) * N_ + n) * D_;
    const float* cur  = states + ((size_t)(b * 3 + 1) * N_ + n) * D_;
    const float* nxt  = states + ((size_t)(b * 3 + 2) * N_ + n) * D_;
#pragma unroll
    for (int k = 0; k < 8; ++k) { y[50 + k] = prev[k]; y[58 + k] = cur[k]; }

    float h[50], h2[50];
    for (int j = 0; j < 50; ++j) {
        float s = b0[j];
        for (int k = 0; k < 66; ++k) s += y[k] * W0[k * 50 + j];
        h[j] = relu_f(s);
    }
    const float* Ws[3] = { W1, W2, W3 };
    const float* bs[3] = { b1, b2, b3 };
    for (int l = 0; l < 3; ++l) {
        for (int j = 0; j < 50; ++j) {
            float s = bs[l][j];
            for (int k = 0; k < 50; ++k) s += h[k] * Ws[l][k * 50 + j];
            h2[j] = relu_f(s);
        }
        for (int j = 0; j < 50; ++j) h[j] = h2[j];
    }
    float* pred = out;
    float* targ = out + B_ * N_ * 2;
#pragma unroll
    for (int v = 0; v < 2; ++v) {
        float s = b4[v];
        for (int k = 0; k < 50; ++k) s += h[k] * W4[k * 2 + v];
        pred[node * 2 + v] = cur[2 + v] + s;
        targ[node * 2 + v] = nxt[2 + v];
    }
}

// ---------------------------------------------------------------------------
extern "C" void kernel_launch(void* const* d_in, const int* in_sizes, int n_in,
                              void* d_out, int out_size, void* d_ws, size_t ws_size,
                              hipStream_t stream)
{
    const float* states = (const float*)d_in[0];
    const float* Wp     = (const float*)d_in[1];
    const float* eW0    = (const float*)d_in[2];
    const float* eW1    = (const float*)d_in[3];
    const float* eW2    = (const float*)d_in[4];
    const float* eW3    = (const float*)d_in[5];
    const float* eW4    = (const float*)d_in[6];
    const float* dW0    = (const float*)d_in[7];
    const float* db0    = (const float*)d_in[8];
    const float* dW1    = (const float*)d_in[9];
    const float* db1    = (const float*)d_in[10];
    const float* dW2    = (const float*)d_in[11];
    const float* db2    = (const float*)d_in[12];
    const float* dW3    = (const float*)d_in[13];
    const float* db3    = (const float*)d_in[14];
    const float* dW4    = (const float*)d_in[15];
    const float* db4    = (const float*)d_in[16];
    float* out = (float*)d_out;

    // workspace layout
    char* ws = (char*)d_ws;
    _Float16* wfrag = (_Float16*)ws;                               //  32768 B
    float* gf       = (float*)(ws + 32768);                        // 409600 B
    float* gc       = (float*)(ws + 32768 + 409600);               // 409600 B
    float* partial  = (float*)(ws + 32768 + 2 * 409600);           // 128*8*16*50*4 B

    pack_weights_kernel<<<4, 256, 0, stream>>>(eW1, eW2, eW3, eW4, wfrag);
    node_proj_kernel<<<8, 256, 0, stream>>>(states, Wp, eW0, gf, gc);
    pair_enc_kernel<<<128, 256, 0, stream>>>(states, gf, gc, wfrag, partial);
    decoder_kernel<<<8, 256, 0, stream>>>(states, partial,
                                          dW0, db0, dW1, db1, dW2, db2,
                                          dW3, db3, dW4, db4, out);
}
